// DVGMatchModule_47923245088975
// MI455X (gfx1250) — matrix operand order repaired and checked by reading, not compiled, not run on hardware
//
#include <hip/hip_runtime.h>
#include <hip/hip_bf16.h>

typedef __attribute__((ext_vector_type(16))) _Float16 v16h;
typedef __attribute__((ext_vector_type(8)))  float    v8f;

#define ATT_SCALE 0.17677669529663687f  // 1/sqrt(32)

// ---------------- WMMA helpers (CDNA5 v_wmma_f32_16x16x32_f16) ----------------

__device__ inline v8f wmma_f16(v16h a, v16h b, v8f c) {
  return __builtin_amdgcn_wmma_f32_16x16x32_f16(
      /*neg_a=*/false, a, /*neg_b=*/false, b,
      /*c_mod=*/(short)0, c, /*reuse_a=*/false, /*reuse_b=*/false);
}

// A tile (16 x 32, MxK), row-major source with leading dim ld.
// lane 0-15: M=lane, elems 0..15 -> K 0..15
// lane16-31: M=lane-16, elems 0..15 -> K 16..31
template<typename T>
__device__ inline v16h load_a_tile(const T* X, int ld, int row0, int k0, int lane) {
  int m = lane & 15;
  int g = (lane >> 4) & 1;
  const T* p = X + (size_t)(row0 + m) * ld + k0;
  v16h a;
#pragma unroll
  for (int e = 0; e < 8; ++e) a[e] = (_Float16)p[g * 16 + e];
#pragma unroll
  for (int e = 0; e < 8; ++e) a[8 + e] = (_Float16)p[g * 16 + 8 + e];
  return a;
}

// B tile for X @ W^T: B[k][n] = W[col0+n][k0+k], W row-major (ld = Cin).
// lane 0-15: N=lane, elems = K 0..15 ; lane16-31: N=lane-16, elems = K 16..31
template<typename T>
__device__ inline v16h load_bT_tile(const T* W, int ld, int col0, int k0, int lane) {
  int n = lane & 15;
  int g = (lane >> 4) & 1;
  const T* p = W + (size_t)(col0 + n) * ld + k0 + g * 16;
  v16h b;
#pragma unroll
  for (int e = 0; e < 16; ++e) b[e] = (_Float16)p[e];
  return b;
}

// B tile non-transposed: B[k][n] = V[k0+k][col0+n], V row-major (ld).
template<typename T>
__device__ inline v16h load_b_tile(const T* V, int ld, int col0, int k0, int lane) {
  int n = lane & 15;
  int g = (lane >> 4) & 1;
  v16h b;
#pragma unroll
  for (int e = 0; e < 16; ++e)
    b[e] = (_Float16)V[(size_t)(k0 + g * 16 + e) * ld + col0 + n];
  return b;
}

// ---------------- distance bias ----------------

__global__ void colsum_kernel(const float* __restrict__ center, float* __restrict__ colsum) {
  __shared__ float cx[256], cy[256], cz[256];
  int b = blockIdx.x, t = threadIdx.x;
  const float* cb = center + (size_t)b * 256 * 3;
  cx[t] = cb[t * 3]; cy[t] = cb[t * 3 + 1]; cz[t] = cb[t * 3 + 2];
  __syncthreads();
  float x = cx[t], y = cy[t], z = cz[t];
  float s = 0.f;
  for (int i = 0; i < 256; ++i) {
    float dx = cx[i] - x, dy = cy[i] - y, dz = cz[i] - z;
    float d = sqrtf(dx * dx + dy * dy + dz * dz);
    s += 1.0f / (d + 0.01f);
  }
  colsum[b * 256 + t] = s;   // sum over i for column j=t (reference sums axis=2)
}

__global__ void distw_kernel(const float* __restrict__ center, const float* __restrict__ colsum,
                             float* __restrict__ distw) {
  int idx = blockIdx.x * 256 + threadIdx.x;       // 16*256*256 total
  int j = idx & 255;
  int i = (idx >> 8) & 255;
  int b = idx >> 16;
  const float* cb = center + (size_t)b * 768;
  float dx = cb[j * 3] - cb[i * 3];
  float dy = cb[j * 3 + 1] - cb[i * 3 + 1];
  float dz = cb[j * 3 + 2] - cb[i * 3 + 2];
  float d = sqrtf(dx * dx + dy * dy + dz * dz);
  float w0 = (1.0f / (d + 0.01f)) / colsum[b * 256 + j];
  size_t base = ((size_t)b * 4 * 256 + i) * 256 + j;
  const size_t hs = 256u * 256u;
  distw[base] = w0;
  distw[base + hs] = -d;
  distw[base + 2 * hs] = 0.f;
  distw[base + 3 * hs] = 0.f;
}

// ---------------- fused features_concat: fc1 + BN + PReLU + fc2 ----------------

__global__ void fc_fused_kernel(const float* __restrict__ detr, const float* __restrict__ W1,
                                const float* __restrict__ b1,
                                const float* __restrict__ bng, const float* __restrict__ bnb,
                                const float* __restrict__ bnm, const float* __restrict__ bnv,
                                const float* __restrict__ pa,
                                const float* __restrict__ W2, const float* __restrict__ b2,
                                float* __restrict__ Y) {
  __shared__ _Float16 As[16 * 1152];
  __shared__ _Float16 Xs[16 * 128];
  int row0 = blockIdx.x * 16;
  int t = threadIdx.x;
  for (int idx = t; idx < 16 * 1152; idx += 256)
    As[idx] = (_Float16)detr[(size_t)(row0 + (idx / 1152)) * 1152 + (idx % 1152)];
  __syncthreads();
  int lane = t & 31, wave = t >> 5;
  int col0 = wave * 16;
  v8f acc = {};
  for (int kc = 0; kc < 36; ++kc) {
    v16h a = load_a_tile(As, 1152, 0, kc * 32, lane);
    v16h b = load_bT_tile(W1, 1152, col0, kc * 32, lane);
    acc = wmma_f16(a, b, acc);
  }
  int n = lane & 15, g = (lane >> 4) & 1;
  int c = col0 + n;
  float gs = bng[c] * rsqrtf(bnv[c] + 1e-5f);
  float mn = bnm[c], bb = bnb[c], al = pa[c], bv = b1[c];
#pragma unroll
  for (int r = 0; r < 8; ++r) {
    float v = acc[r] + bv;
    v = (v - mn) * gs + bb;
    v = v >= 0.f ? v : al * v;
    Xs[(r + g * 8) * 128 + c] = (_Float16)v;
  }
  __syncthreads();
  v8f acc2 = {};
#pragma unroll
  for (int kc = 0; kc < 4; ++kc) {
    v16h a = load_a_tile(Xs, 128, 0, kc * 32, lane);
    v16h b = load_bT_tile(W2, 128, col0, kc * 32, lane);
    acc2 = wmma_f16(a, b, acc2);
  }
  float bv2 = b2[c];
#pragma unroll
  for (int r = 0; r < 8; ++r)
    Y[(size_t)(row0 + r + g * 8) * 128 + c] = acc2[r] + bv2;
}

// ---------------- generic 128-col GEMM:  Y = X @ W^T + bias ----------------
// Block = 8 waves, 16 rows x 128 cols per block; wave w -> col tile w. In-place safe.

template<typename IN_T, typename OUT_T>
__global__ void gemm128_kernel(const IN_T* __restrict__ X, const float* __restrict__ W,
                               const float* __restrict__ bias, OUT_T* __restrict__ Y) {
  __shared__ _Float16 As[16 * 128];
  int row0 = blockIdx.x * 16;
  int t = threadIdx.x;
  for (int idx = t; idx < 16 * 128; idx += 256)
    As[idx] = (_Float16)X[(size_t)(row0 + (idx >> 7)) * 128 + (idx & 127)];
  __syncthreads();
  int lane = t & 31, wave = t >> 5;
  int col0 = wave * 16;
  v8f acc = {};
#pragma unroll
  for (int kc = 0; kc < 4; ++kc) {
    v16h a = load_a_tile(As, 128, 0, kc * 32, lane);
    v16h b = load_bT_tile(W, 128, col0, kc * 32, lane);
    acc = wmma_f16(a, b, acc);
  }
  int n = lane & 15, g = (lane >> 4) & 1;
  float bv = bias ? bias[col0 + n] : 0.f;
#pragma unroll
  for (int r = 0; r < 8; ++r)
    Y[(size_t)(row0 + r + g * 8) * 128 + col0 + n] = (OUT_T)(acc[r] + bv);
}

// ---------------- fused attention: S = QK^T*s (+bias|mask) ; softmax ; P@V ----------------
// One wave per (batch, head, 16-query tile). DK=32 -> one WMMA per score tile.

__global__ void attention_kernel(const _Float16* __restrict__ Q, const _Float16* __restrict__ Kb,
                                 const _Float16* __restrict__ Vb, _Float16* __restrict__ Out,
                                 int Nq, int Nk,
                                 const float* __restrict__ bias, int bias_div,
                                 const unsigned char* __restrict__ mask) {
  __shared__ float    S[16 * 256];
  __shared__ _Float16 P[16 * 256];
  int ntq = Nq >> 4;
  int qt = blockIdx.x % ntq;
  int h  = (blockIdx.x / ntq) & 3;
  int b  = blockIdx.x / (ntq * 4);
  int lane = threadIdx.x;
  int q0 = qt * 16;
  const _Float16* Qb  = Q  + (size_t)b * Nq * 128 + h * 32;
  const _Float16* Kbb = Kb + (size_t)b * Nk * 128 + h * 32;
  const _Float16* Vbb = Vb + (size_t)b * Nk * 128 + h * 32;

  v16h a = load_a_tile(Qb, 128, q0, 0, lane);        // 16 x 32 query tile
  int n = lane & 15, g = (lane >> 4) & 1;
  const float* brow = bias ? bias + (((size_t)(b / bias_div) * 4 + h) * 256) * 256 : nullptr;

  for (int kt = 0; kt < Nk; kt += 16) {
    v16h bk = load_bT_tile(Kbb, 128, kt, 0, lane);   // B[d][key] = K[key][d]
    v8f c = {};
    c = wmma_f16(a, bk, c);
    int kk = kt + n;
    bool mk = (mask != nullptr) && (mask[(size_t)b * Nk + kk] != 0);
#pragma unroll
    for (int r = 0; r < 8; ++r) {
      int q = r + g * 8;
      float s = c[r] * ATT_SCALE;
      if (brow) s += brow[(size_t)(q0 + q) * 256 + kk];
      if (mk) s = -1e9f;
      S[q * 256 + kk] = s;
    }
  }
  __syncthreads();
  {
    int row = lane & 15, half = lane >> 4;
    float* Srow = S + row * 256;
    int c0 = half * (Nk >> 1), c1 = c0 + (Nk >> 1);
    float mx = -3.4e38f;
    for (int c = c0; c < c1; ++c) mx = fmaxf(mx, Srow[c]);
    mx = fmaxf(mx, __shfl_xor(mx, 16, 32));
    float sum = 0.f;
    for (int c = c0; c < c1; ++c) { float e = __expf(Srow[c] - mx); Srow[c] = e; sum += e; }
    sum += __shfl_xor(sum, 16, 32);
    float inv = 1.0f / sum;
    _Float16* Prow = P + row * 256;
    for (int c = c0; c < c1; ++c) Prow[c] = (_Float16)(Srow[c] * inv);
  }
  __syncthreads();
  v8f o0 = {}, o1 = {};
  for (int kc = 0; kc < Nk; kc += 32) {
    v16h ap = load_a_tile(P, 256, 0, kc, lane);
    v16h v0 = load_b_tile(Vbb, 128, 0,  kc, lane);
    v16h v1 = load_b_tile(Vbb, 128, 16, kc, lane);
    o0 = wmma_f16(ap, v0, o0);
    o1 = wmma_f16(ap, v1, o1);
  }
  _Float16* Ob = Out + (size_t)b * Nq * 128 + h * 32;
#pragma unroll
  for (int r = 0; r < 8; ++r) {
    int q = q0 + r + g * 8;
    Ob[(size_t)q * 128 + n]      = (_Float16)o0[r];
    Ob[(size_t)q * 128 + 16 + n] = (_Float16)o1[r];
  }
}

// ---------------- residual + LayerNorm (one block per 128-wide row) ----------------

__global__ void add_ln_kernel(const float* __restrict__ resid, const float* __restrict__ o,
                              const float* __restrict__ gma, const float* __restrict__ bta,
                              float* __restrict__ Y) {
  __shared__ float red[128];
  int row = blockIdx.x, t = threadIdx.x;
  float v = resid[(size_t)row * 128 + t] + o[(size_t)row * 128 + t];
  red[t] = v; __syncthreads();
  for (int s = 64; s > 0; s >>= 1) { if (t < s) red[t] += red[t + s]; __syncthreads(); }
  float mu = red[0] * (1.0f / 128.f);
  __syncthreads();
  float d = v - mu;
  red[t] = d * d; __syncthreads();
  for (int s = 64; s > 0; s >>= 1) { if (t < s) red[t] += red[t + s]; __syncthreads(); }
  float var = red[0] * (1.0f / 128.f);
  Y[(size_t)row * 128 + t] = d * rsqrtf(var + 1e-5f) * gma[t] + bta[t];
}

// ---------------- misc elementwise ----------------

__global__ void broadcast_kernel(const float* __restrict__ src, float* __restrict__ dst) {
  size_t idx = (size_t)blockIdx.x * 256 + threadIdx.x;   // 256*256*128 total
  size_t inner = idx % (256u * 128u);
  size_t bm = idx / (256u * 128u);
  dst[idx] = src[(bm >> 4) * (256u * 128u) + inner];
}

__global__ void bn_prelu_kernel(float* __restrict__ Y, const float* __restrict__ g,
                                const float* __restrict__ b, const float* __restrict__ m,
                                const float* __restrict__ v, const float* __restrict__ a,
                                int a_scalar) {
  size_t idx = (size_t)blockIdx.x * 256 + threadIdx.x;
  int c = idx & 127;
  float x = Y[idx];
  x = (x - m[c]) * (g[c] * rsqrtf(v[c] + 1e-5f)) + b[c];
  float al = a_scalar ? a[0] : a[c];
  Y[idx] = x >= 0.f ? x : al * x;
}

__global__ void dot_kernel(const float* __restrict__ Y, const float* __restrict__ w3,
                           const float* __restrict__ b3, float* __restrict__ out) {
  __shared__ float red[128];
  int row = blockIdx.x, t = threadIdx.x;
  red[t] = Y[(size_t)row * 128 + t] * w3[t];
  __syncthreads();
  for (int s = 64; s > 0; s >>= 1) { if (t < s) red[t] += red[t + s]; __syncthreads(); }
  if (t == 0) out[row] = red[0] + b3[0];
}

// ---------------- orchestration ----------------

extern "C" void kernel_launch(void* const* d_in, const int* in_sizes, int n_in,
                              void* d_out, int out_size, void* d_ws, size_t ws_size,
                              hipStream_t stream) {
  (void)in_sizes; (void)n_in; (void)out_size; (void)ws_size;
  const float* center = (const float*)d_in[0];
  const float* detr   = (const float*)d_in[1];
  const float* lang   = (const float*)d_in[2];
  const float* fc_W1  = (const float*)d_in[3];
  const float* fc_b1  = (const float*)d_in[4];
  const float* bn1_g  = (const float*)d_in[5];
  const float* bn1_b  = (const float*)d_in[6];
  const float* bn1_m  = (const float*)d_in[7];
  const float* bn1_v  = (const float*)d_in[8];
  const float* mbn1_g = (const float*)d_in[9];
  const float* mbn1_b = (const float*)d_in[10];
  const float* mbn1_m = (const float*)d_in[11];
  const float* mbn1_v = (const float*)d_in[12];
  const float* mbn2_g = (const float*)d_in[13];
  const float* mbn2_b = (const float*)d_in[14];
  const float* mbn2_m = (const float*)d_in[15];
  const float* mbn2_v = (const float*)d_in[16];
  const float* pr1_a  = (const float*)d_in[17];
  const float* fc_W2  = (const float*)d_in[18];
  const float* fc_b2  = (const float*)d_in[19];
  const float* sa_Wq  = (const float*)d_in[20];
  const float* sa_Wk  = (const float*)d_in[21];
  const float* sa_Wv  = (const float*)d_in[22];
  const float* sa_Wo  = (const float*)d_in[23];
  const float* sa_bq  = (const float*)d_in[24];
  const float* sa_bk  = (const float*)d_in[25];
  const float* sa_bv  = (const float*)d_in[26];
  const float* sa_bo  = (const float*)d_in[27];
  const float* sa_lb  = (const float*)d_in[28];
  const float* sa_lg  = (const float*)d_in[29];
  const float* ca_Wq  = (const float*)d_in[30];
  const float* ca_Wk  = (const float*)d_in[31];
  const float* ca_Wv  = (const float*)d_in[32];
  const float* ca_Wo  = (const float*)d_in[33];
  const float* ca_bq  = (const float*)d_in[34];
  const float* ca_bk  = (const float*)d_in[35];
  const float* ca_bv  = (const float*)d_in[36];
  const float* ca_bo  = (const float*)d_in[37];
  const float* ca_lb  = (const float*)d_in[38];
  const float* ca_lg  = (const float*)d_in[39];
  const float* m_W1   = (const float*)d_in[40];
  const float* m_b1   = (const float*)d_in[41];
  const float* mp1_a  = (const float*)d_in[42];
  const float* m_W2   = (const float*)d_in[43];
  const float* m_b2   = (const float*)d_in[44];
  const float* mp2_a  = (const float*)d_in[45];
  const float* m_W3   = (const float*)d_in[46];
  const float* m_b3   = (const float*)d_in[47];
  const unsigned char* amask = (const unsigned char*)d_in[48];

  // workspace carve
  char* w = (char*)d_ws;
  auto carve = [&](size_t bytes) {
    char* p = w; w += (bytes + 255) & ~(size_t)255; return p;
  };
  float*    distw  = (float*)   carve((size_t)16 * 4 * 256 * 256 * 4);
  float*    colsum = (float*)   carve((size_t)16 * 256 * 4);
  float*    xsmall = (float*)   carve((size_t)4096 * 128 * 4);
  float*    fsmall = (float*)   carve((size_t)4096 * 128 * 4);
  _Float16* qs     = (_Float16*)carve((size_t)4096 * 128 * 2);
  _Float16* ks     = (_Float16*)carve((size_t)4096 * 128 * 2);
  _Float16* vs     = (_Float16*)carve((size_t)4096 * 128 * 2);
  _Float16* cs     = (_Float16*)carve((size_t)4096 * 128 * 2);
  float*    otmps  = (float*)   carve((size_t)4096 * 128 * 4);
  float*    A      = (float*)   carve((size_t)65536 * 128 * 4);
  float*    otmp   = (float*)   carve((size_t)65536 * 128 * 4);
  _Float16* Qh     = (_Float16*)carve((size_t)65536 * 128 * 2);
  _Float16* Kh     = (_Float16*)carve((size_t)65536 * 128 * 2);
  _Float16* Vh     = (_Float16*)carve((size_t)65536 * 128 * 2);
  _Float16* Ch     = (_Float16*)carve((size_t)65536 * 128 * 2);

  // 1) distance bias (B,4,K,K)
  colsum_kernel<<<16, 256, 0, stream>>>(center, colsum);
  distw_kernel<<<(16 * 256 * 256) / 256, 256, 0, stream>>>(center, colsum, distw);

  // 2) features_concat (fused fc1+BN+PReLU+fc2) -> xsmall (16*256, 128)
  fc_fused_kernel<<<4096 / 16, 256, 0, stream>>>(detr, fc_W1, fc_b1, bn1_g, bn1_b, bn1_m,
                                                 bn1_v, pr1_a, fc_W2, fc_b2, xsmall);

  // 3) self-attention layer 0 on (16, 256, 128)
  gemm128_kernel<float, _Float16><<<256, 256, 0, stream>>>(xsmall, sa_Wq, sa_bq, qs);
  gemm128_kernel<float, _Float16><<<256, 256, 0, stream>>>(xsmall, sa_Wk, sa_bk, ks);
  gemm128_kernel<float, _Float16><<<256, 256, 0, stream>>>(xsmall, sa_Wv, sa_bv, vs);
  attention_kernel<<<16 * 4 * 16, 32, 0, stream>>>(qs, ks, vs, cs, 256, 256, distw, 1, nullptr);
  gemm128_kernel<_Float16, float><<<256, 256, 0, stream>>>(cs, sa_Wo, sa_bo, otmps);
  add_ln_kernel<<<4096, 128, 0, stream>>>(xsmall, otmps, sa_lg, sa_lb, fsmall);

  // 4) broadcast to (B*M, K, HID) = (256, 256, 128)
  broadcast_kernel<<<(256 * 256 * 128) / 256, 256, 0, stream>>>(fsmall, A);

  // 5) [SA(i) for i>=1] + CA(i) stack
  for (int layer = 0; layer < 2; ++layer) {
    size_t woff = (size_t)layer * 128 * 128;
    size_t boff = (size_t)layer * 128;
    if (layer > 0) {
      gemm128_kernel<float, _Float16><<<4096, 256, 0, stream>>>(A, sa_Wq + woff, sa_bq + boff, Qh);
      gemm128_kernel<float, _Float16><<<4096, 256, 0, stream>>>(A, sa_Wk + woff, sa_bk + boff, Kh);
      gemm128_kernel<float, _Float16><<<4096, 256, 0, stream>>>(A, sa_Wv + woff, sa_bv + boff, Vh);
      attention_kernel<<<256 * 4 * 16, 32, 0, stream>>>(Qh, Kh, Vh, Ch, 256, 256,
                                                        distw, 16, nullptr);
      gemm128_kernel<_Float16, float><<<4096, 256, 0, stream>>>(Ch, sa_Wo + woff, sa_bo + boff, otmp);
      add_ln_kernel<<<65536, 128, 0, stream>>>(A, otmp, sa_lg + boff, sa_lb + boff, A);
    }
    gemm128_kernel<float, _Float16><<<4096, 256, 0, stream>>>(A, ca_Wq + woff, ca_bq + boff, Qh);
    gemm128_kernel<float, _Float16><<<1024, 256, 0, stream>>>(lang, ca_Wk + woff, ca_bk + boff, Kh);
    gemm128_kernel<float, _Float16><<<1024, 256, 0, stream>>>(lang, ca_Wv + woff, ca_bv + boff, Vh);
    attention_kernel<<<256 * 4 * 16, 32, 0, stream>>>(Qh, Kh, Vh, Ch, 256, 64,
                                                      nullptr, 1, amask);
    gemm128_kernel<_Float16, float><<<4096, 256, 0, stream>>>(Ch, ca_Wo + woff, ca_bo + boff, otmp);
    add_ln_kernel<<<65536, 128, 0, stream>>>(A, otmp, ca_lg + boff, ca_lb + boff, A);
  }

  // 6) match head
  gemm128_kernel<float, float><<<4096, 256, 0, stream>>>(A, m_W1, m_b1, otmp);
  bn_prelu_kernel<<<(65536 * 128) / 256, 256, 0, stream>>>(otmp, mbn1_g, mbn1_b, mbn1_m,
                                                           mbn1_v, mp1_a, 1);
  gemm128_kernel<float, float><<<4096, 256, 0, stream>>>(otmp, m_W2, m_b2, otmp);  // in-place safe
  bn_prelu_kernel<<<(65536 * 128) / 256, 256, 0, stream>>>(otmp, mbn2_g, mbn2_b, mbn2_m,
                                                           mbn2_v, mp2_a, 1);
  dot_kernel<<<65536, 128, 0, stream>>>(otmp, m_W3, m_b3, (float*)d_out);
}
